// CompoundGCN_77489799954649
// MI455X (gfx1250) — matrix operand order repaired and checked by reading, not compiled, not run on hardware
//
#include <hip/hip_runtime.h>
#include <hip/hip_bf16.h>

#define N_NODES 100000
#define N_EDGES 800000
#define N_BATCH 256
#define ND 64
#define ED 64
#define HM 64
#define HD 128
#define OD 16
#define MAX_DEG 10
#define NEG_INF_F (-3.402823466e38f)

typedef __attribute__((ext_vector_type(16))) _Float16 v16h;
typedef __attribute__((ext_vector_type(8)))  float    v8f;

__device__ inline void atomic_max_f32(float* addr, float val) {
    unsigned int* ua = (unsigned int*)addr;
    unsigned int old = *ua;
    while (true) {
        float of = __uint_as_float(old);
        if (of >= val) break;
        unsigned int assumed = old;
        old = atomicCAS(ua, assumed, __float_as_uint(val));
        if (old == assumed) break;
    }
}

// ---------------- K0: initialize scratch + running-max buffer ----------------
__global__ void k_init(float* __restrict__ h, float* __restrict__ nodeagg,
                       float* __restrict__ pooled, int* __restrict__ cnt,
                       float* __restrict__ agg) {
    long idx = (long)blockIdx.x * blockDim.x + threadIdx.x;
    if (idx < (long)N_EDGES * HM) agg[idx] = NEG_INF_F;      // emb_edge max identity
    if (idx < (long)N_NODES * ND) { h[idx] = 0.f; nodeagg[idx] = 0.f; }
    if (idx < N_NODES)            cnt[idx] = 0;
    if (idx < N_BATCH * HD)       pooled[idx] = 0.f;
}

// ---------------- K1: degree count + neighbor feature sum --------------------
__global__ void k_deg_h(const float* __restrict__ x, const int* __restrict__ row,
                        const int* __restrict__ col, float* __restrict__ h,
                        int* __restrict__ cnt) {
    long idx = (long)blockIdx.x * blockDim.x + threadIdx.x;
    if (idx >= (long)N_EDGES * ND) return;
    int e = (int)(idx >> 6);
    int f = (int)(idx & 63);
    int r = row[e], c = col[e];
    atomicAdd(&h[(long)c * ND + f], x[(long)r * ND + f]);
    if (f == 0) atomicAdd(&cnt[c], 1);
}

// ---------------- K2: MFConv per-degree linears (emb_node) -------------------
__global__ void k_embnode(const float* __restrict__ x, const float* __restrict__ h,
                          const int* __restrict__ cnt,
                          const float* __restrict__ Wl, const float* __restrict__ bl,
                          const float* __restrict__ Wr,
                          float* __restrict__ emb_node_out, float* __restrict__ x_act) {
    int i = blockIdx.x;
    int j = threadIdx.x;                        // 64 threads: output feature
    __shared__ float xs[ND], hs[ND];
    xs[j] = x[(long)i * ND + j];
    hs[j] = h[(long)i * ND + j];
    __syncthreads();
    int d = cnt[i]; if (d > MAX_DEG) d = MAX_DEG;
    const float* wl = Wl + (long)d * ND * HM;
    const float* wr = Wr + (long)d * ND * HM;
    float acc = bl[d * HM + j];
#pragma unroll 8
    for (int k = 0; k < ND; ++k)
        acc += hs[k] * wl[k * HM + j] + xs[k] * wr[k * HM + j];
    emb_node_out[(long)i * HM + j] = acc;
    x_act[(long)i * HM + j] = fmaxf(acc, 0.f);
}

// ---------------- K3: EdgeConv via WMMA + fused atomic-max scatter -----------
// Each wave: 16-edge tile x 64 output cols. A = [xi | xj-xi] (16x128 f16),
// B = We (128x64) pre-swizzled in LDS into per-(kchunk,ntile,lane) v16h
// fragments so every WMMA B operand is a single contiguous 32B LDS load
// (2x ds_load_b128) instead of 16 scalar ds_load_u16.
__global__ void __launch_bounds__(256)
k_edgeconv(const float* __restrict__ edge_attr, const int* __restrict__ row,
           const int* __restrict__ col, const float* __restrict__ We,
           const float* __restrict__ be, float* __restrict__ agg) {
    // [kchunk][ntile][lane][hh] -> We[(c*32 + (hh<8 ? (lane>>4)*8 + hh : 16 + (lane>>4)*8 + hh-8))*64 + t*16 + (lane&15)]
    __shared__ __align__(32) _Float16 sWeB[4][4][32][16];   // 16 KB
    for (int idx = threadIdx.x; idx < 4 * 4 * 32 * 16; idx += blockDim.x) {
        int hh = idx & 15;
        int ln = (idx >> 4) & 31;
        int t  = (idx >> 9) & 3;
        int c  = idx >> 11;
        int k    = c * 32 + ((hh < 8) ? ((ln >> 4) * 8 + hh) : (16 + (ln >> 4) * 8 + (hh - 8)));
        int ncol = t * 16 + (ln & 15);
        sWeB[c][t][ln][hh] = (_Float16)We[k * HM + ncol];
    }
    __syncthreads();

    const int wave  = threadIdx.x >> 5;
    const int lane  = threadIdx.x & 31;
    const int tile  = blockIdx.x * 8 + wave;
    const int ebase = tile * 16;
    const int m     = lane & 15;                // edge row (M) for A
    const int hi    = lane >> 4;                // lane-group (K split)

    const int eM = ebase + m;
    const float* xi = edge_attr + (long)col[eM] * ED;   // x_i = edge_attr[col]
    const float* xj = edge_attr + (long)row[eM] * ED;   // x_j = edge_attr[row]

    v8f acc[4];
    const v8f vzero = {0.f, 0.f, 0.f, 0.f, 0.f, 0.f, 0.f, 0.f};
#pragma unroll
    for (int t = 0; t < 4; ++t) acc[t] = vzero;

    const int kbaseA = hi * 8;                  // A layout: lanes 16-31 hold K+8 block

#pragma unroll
    for (int c = 0; c < 4; ++c) {               // K chunks of 32 (total K=128)
        const int K0 = c * 32;
        v16h a;
#pragma unroll
        for (int hh = 0; hh < 16; ++hh) {
            int k = K0 + ((hh < 8) ? (kbaseA + hh) : (16 + kbaseA + (hh - 8)));
            float v = (k < ED) ? xi[k] : (xj[k - ED] - xi[k - ED]);
            a[hh] = (_Float16)v;
        }
#pragma unroll
        for (int t = 0; t < 4; ++t) {           // 4 N-tiles of 16 cols
            const v16h b = *(const v16h*)&sWeB[c][t][lane][0];
            acc[t] = __builtin_amdgcn_wmma_f32_16x16x32_f16(
                false, a, false, b, (short)0, acc[t], false, false);
        }
    }

    // Fused bias + segment-max scatter: D layout lane L: N=L%16, VGPR r -> M=r+8*(L/16)
#pragma unroll
    for (int t = 0; t < 4; ++t) {
        const int j = t * 16 + m;
        const float bias = be[j];
#pragma unroll
        for (int r = 0; r < 8; ++r) {
            const int eidx = ebase + r + 8 * hi;
            const float val = acc[t][r] + bias;
            const int node = col[eidx];
            atomic_max_f32(&agg[(long)node * HM + j], val);
        }
    }
}

// ---------------- K4: finalize emb_edge (empty segments -> 0) + relu ---------
__global__ void k_finalize(const int* __restrict__ cnt, float* __restrict__ emb_edge_out,
                           float* __restrict__ e_act) {
    long idx = (long)blockIdx.x * blockDim.x + threadIdx.x;
    if (idx >= (long)N_EDGES * HM) return;
    long i = idx >> 6;
    float v = emb_edge_out[idx];
    bool has = (i < N_NODES) && (cnt[i] > 0);
    v = has ? v : 0.f;
    emb_edge_out[idx] = v;
    if (i < N_NODES) e_act[idx] = fmaxf(v, 0.f);
}

// ---------------- K5: scatter x_act half of pe into node_agg -----------------
__global__ void k_pescatter(const int* __restrict__ row, const int* __restrict__ col,
                            const float* __restrict__ x_act, float* __restrict__ nodeagg) {
    long idx = (long)blockIdx.x * blockDim.x + threadIdx.x;
    if (idx >= (long)N_EDGES * HM) return;
    int e = (int)(idx >> 6);
    int f = (int)(idx & 63);
    atomicAdd(&nodeagg[(long)col[e] * HM + f], x_act[(long)row[e] * HM + f]);
}

// ---------------- K6: global add pool (second half = cnt * e_act) ------------
__global__ void k_pool(const int* __restrict__ batch, const int* __restrict__ cnt,
                       const float* __restrict__ nodeagg, const float* __restrict__ e_act,
                       float* __restrict__ pooled) {
    long idx = (long)blockIdx.x * blockDim.x + threadIdx.x;
    if (idx >= (long)N_NODES * HD) return;
    int i = (int)(idx >> 7);
    int f = (int)(idx & 127);
    int b = batch[i];
    float v = (f < HM) ? nodeagg[(long)i * HM + f]
                       : ((float)cnt[i]) * e_act[(long)i * HM + (f - HM)];
    atomicAdd(&pooled[b * HD + f], v);
}

// ---------------- K7/K8: MLP head --------------------------------------------
__global__ void k_head1(const float* __restrict__ pooled, const float* __restrict__ W1,
                        const float* __restrict__ b1, float* __restrict__ hid) {
    int idx = blockIdx.x * blockDim.x + threadIdx.x;
    if (idx >= N_BATCH * HD) return;
    int b = idx / HD, j = idx % HD;
    float acc = b1[j];
#pragma unroll 8
    for (int k = 0; k < 2 * HM; ++k) acc += pooled[b * HD + k] * W1[k * HD + j];
    hid[idx] = acc;
}

__global__ void k_head2(const float* __restrict__ hid, const float* __restrict__ W2,
                        const float* __restrict__ b2, float* __restrict__ out) {
    int idx = blockIdx.x * blockDim.x + threadIdx.x;
    if (idx >= N_BATCH * OD) return;
    int b = idx / OD, o = idx % OD;
    float acc = b2[o];
#pragma unroll 8
    for (int k = 0; k < HD; ++k) acc += hid[b * HD + k] * W2[k * OD + o];
    out[idx] = acc;
}

extern "C" void kernel_launch(void* const* d_in, const int* in_sizes, int n_in,
                              void* d_out, int out_size, void* d_ws, size_t ws_size,
                              hipStream_t stream) {
    const float* x         = (const float*)d_in[0];
    const float* edge_attr = (const float*)d_in[1];
    const int*   edge_index= (const int*)d_in[2];
    const int*   batch     = (const int*)d_in[3];
    const float* Wl        = (const float*)d_in[4];
    const float* bl        = (const float*)d_in[5];
    const float* Wr        = (const float*)d_in[6];
    const float* We        = (const float*)d_in[7];
    const float* be        = (const float*)d_in[8];
    const float* W1        = (const float*)d_in[9];
    const float* b1        = (const float*)d_in[10];
    const float* W2        = (const float*)d_in[11];
    const float* b2        = (const float*)d_in[12];

    const int* row = edge_index;             // edge_index[0]
    const int* col = edge_index + N_EDGES;   // edge_index[1]

    // Output tuple layout: out [B*OD] | emb_node [N*HM] | emb_edge [E*HM]
    float* out          = (float*)d_out;
    float* emb_node_out = out + N_BATCH * OD;
    float* emb_edge_out = emb_node_out + (long)N_NODES * HM;

    // Workspace carve-up (~103 MB)
    float* ws      = (float*)d_ws;
    float* h       = ws;                               // N*64
    float* x_act   = h + (long)N_NODES * ND;           // N*64
    float* e_act   = x_act + (long)N_NODES * HM;       // N*64
    float* nodeagg = e_act + (long)N_NODES * HM;       // N*64
    float* pooled  = nodeagg + (long)N_NODES * HM;     // B*128
    float* hid     = pooled + N_BATCH * HD;            // B*128
    int*   cnt     = (int*)(hid + N_BATCH * HD);       // N

    const int tb = 256;
    const long eh = (long)N_EDGES * HM;                // 51.2M

    k_init     <<<(unsigned)((eh + tb - 1) / tb), tb, 0, stream>>>(h, nodeagg, pooled, cnt, emb_edge_out);
    k_deg_h    <<<(unsigned)(((long)N_EDGES * ND + tb - 1) / tb), tb, 0, stream>>>(x, row, col, h, cnt);
    k_embnode  <<<N_NODES, ND, 0, stream>>>(x, h, cnt, Wl, bl, Wr, emb_node_out, x_act);
    k_edgeconv <<<N_EDGES / 16 / 8, 256, 0, stream>>>(edge_attr, row, col, We, be, emb_edge_out);
    k_finalize <<<(unsigned)((eh + tb - 1) / tb), tb, 0, stream>>>(cnt, emb_edge_out, e_act);
    k_pescatter<<<(unsigned)((eh + tb - 1) / tb), tb, 0, stream>>>(row, col, x_act, nodeagg);
    k_pool     <<<(unsigned)(((long)N_NODES * HD + tb - 1) / tb), tb, 0, stream>>>(batch, cnt, nodeagg, e_act, pooled);
    k_head1    <<<(N_BATCH * HD + tb - 1) / tb, tb, 0, stream>>>(pooled, W1, b1, hid);
    k_head2    <<<(N_BATCH * OD + tb - 1) / tb, tb, 0, stream>>>(hid, W2, b2, out);
}
